// ScaledDotProductAttention_68143951118741
// MI455X (gfx1250) — matrix operand order repaired and checked by reading, not compiled, not run on hardware
//
#include <hip/hip_runtime.h>

typedef __attribute__((ext_vector_type(16))) _Float16 v16h;
typedef __attribute__((ext_vector_type(8)))  float    v8f;

#define NB 2
#define NH 16
#define NS 2048
#define ND 64
#define BQ 64          // q rows per workgroup (16 per wave, 4 waves)
#define BK 64          // key chunk
#define LSTR 72        // padded LDS row stride (halves) to avoid bank conflicts
#define NCHUNK (NS / BK)

static __device__ __forceinline__ v8f wmma16(v16h a, v16h b, v8f c) {
  // D = A(16x32 f16) * B(32x16 f16) + C(16x16 f32)
  return __builtin_amdgcn_wmma_f32_16x16x32_f16(false, a, false, b, (short)0, c,
                                                false, false);
}

__global__ __launch_bounds__(128)
void attn_fwd(const float* __restrict__ Q, const float* __restrict__ K,
              const float* __restrict__ V, const float* __restrict__ Mk,
              const int* __restrict__ MaskNow,
              float* __restrict__ Out, float* __restrict__ Probs)
{
  __shared__ _Float16 Qs[BQ * LSTR];
  __shared__ _Float16 Ks[BK * LSTR];
  __shared__ _Float16 Vs[BK * LSTR];
  __shared__ _Float16 Ps[BQ * LSTR];   // wave w stages its 16xBK P tile in rows 16w..16w+15

  const int tid  = threadIdx.x;
  const int wave = tid >> 5;
  const int lane = tid & 31;
  const int g    = lane >> 4;          // half-wave group (0/1)
  const int ln   = lane & 15;

  const int q0 = blockIdx.x * BQ;
  const int h  = blockIdx.y;
  const int b  = blockIdx.z;
  const int mn = MaskNow[0];           // 0 -> causal diag=0, 1 -> strict causal
  const float scale = 0.125f;          // 1/sqrt(64)

  const size_t bh = (size_t)b * NH + h;
  const float* Qp = Q + bh * NS * ND;
  const float* Kp = K + bh * NS * ND;
  const float* Vp = V + bh * NS * ND;
  float* Op = Out + bh * NS * ND;
  float* Pp = Probs + bh * (size_t)NS * NS;

  // ---- stage Q tile (f32 global -> f16 LDS), once ----
  for (int i = tid; i < BQ * ND; i += 128) {
    int r = i >> 6, c = i & 63;
    Qs[r * LSTR + c] = (_Float16)Qp[(size_t)(q0 + r) * ND + c];
  }
  __syncthreads();

  // per-lane padding-mask values for rows m = r + 8g of this wave's q tile
  float rowok[8];
  #pragma unroll
  for (int r = 0; r < 8; ++r) {
    int q = q0 + wave * 16 + r + 8 * g;
    rowok[r] = Mk[(size_t)b * NS + q];
  }

  // Q A-fragments (16x32 f16, two K-steps covering d=0..63); ISA 7.12.2 layout:
  // lane holds row m=lane&15; half j -> k = j + (j&8) + 8*(lane>=16)
  v16h aq0, aq1;
  #pragma unroll
  for (int j = 0; j < 16; ++j) {
    int kk = j + (j & 8) + 8 * g;
    aq0[j] = Qs[(wave * 16 + ln) * LSTR + kk];
    aq1[j] = Qs[(wave * 16 + ln) * LSTR + 32 + kk];
  }

  float m_run[8], l_run[8];
  #pragma unroll
  for (int r = 0; r < 8; ++r) { m_run[r] = -3.0e38f; l_run[r] = 0.0f; }

  // prefetch line index for this thread: 64 rows x two 128B halves per chunk
  const int pf_row  = tid >> 1;
  const int pf_half = (tid & 1) * 32;   // floats (=128B) offset within a row

  // ================= pass 1: online row max / row sum =================
  for (int kc = 0; kc < NCHUNK; ++kc) {
    const int k0 = kc * BK;
    __syncthreads();
    for (int i = tid; i < BK * ND; i += 128) {
      int r = i >> 6, c = i & 63;
      Ks[r * LSTR + c] = (_Float16)Kp[(size_t)(k0 + r) * ND + c];
    }
    // warm L2/WGP$ for the next K chunk (global_prefetch_b8)
    if (kc + 1 < NCHUNK)
      __builtin_prefetch(Kp + (size_t)(k0 + BK + pf_row) * ND + pf_half, 0, 3);
    __syncthreads();

    float sbuf[4][8];
    #pragma unroll
    for (int nt = 0; nt < 4; ++nt) {
      // B fragment = K^T (32x16): half j -> kdim = j + (j&8) + 8g, n(key) = nt*16+ln
      v16h b0, b1;
      #pragma unroll
      for (int j = 0; j < 16; ++j) {
        int kk = j + (j & 8) + 8 * g;
        b0[j] = Ks[(nt * 16 + ln) * LSTR + kk];
        b1[j] = Ks[(nt * 16 + ln) * LSTR + 32 + kk];
      }
      v8f c = {0.f,0.f,0.f,0.f,0.f,0.f,0.f,0.f};
      c = wmma16(aq0, b0, c);
      c = wmma16(aq1, b1, c);
      #pragma unroll
      for (int r = 0; r < 8; ++r) {
        int q = q0 + wave * 16 + r + 8 * g;
        int k = k0 + nt * 16 + ln;
        bool ok = ((k + mn) <= q) && (rowok[r] != 0.0f);
        sbuf[nt][r] = ok ? c[r] * scale : -1.0e9f;
      }
    }
    #pragma unroll
    for (int r = 0; r < 8; ++r) {
      float mx = fmaxf(fmaxf(sbuf[0][r], sbuf[1][r]),
                       fmaxf(sbuf[2][r], sbuf[3][r]));
      #pragma unroll
      for (int off = 1; off < 16; off <<= 1)
        mx = fmaxf(mx, __shfl_xor(mx, off, 32));
      float mnew = fmaxf(m_run[r], mx);
      float se = __expf(sbuf[0][r] - mnew) + __expf(sbuf[1][r] - mnew)
               + __expf(sbuf[2][r] - mnew) + __expf(sbuf[3][r] - mnew);
      #pragma unroll
      for (int off = 1; off < 16; off <<= 1)
        se += __shfl_xor(se, off, 32);
      l_run[r] = l_run[r] * __expf(m_run[r] - mnew) + se;
      m_run[r] = mnew;
    }
  }

  float inv_l[8];
  #pragma unroll
  for (int r = 0; r < 8; ++r) inv_l[r] = 1.0f / l_run[r];

  // ================= pass 2: write probs (NT) + accumulate P.V =================
  v8f acc[4];
  const v8f vzero = {0.f,0.f,0.f,0.f,0.f,0.f,0.f,0.f};
  #pragma unroll
  for (int nt = 0; nt < 4; ++nt) acc[nt] = vzero;

  for (int kc = 0; kc < NCHUNK; ++kc) {
    const int k0 = kc * BK;
    __syncthreads();
    for (int i = tid; i < BK * ND; i += 128) {
      int r = i >> 6, c = i & 63;
      Ks[r * LSTR + c] = (_Float16)Kp[(size_t)(k0 + r) * ND + c];
      Vs[r * LSTR + c] = (_Float16)Vp[(size_t)(k0 + r) * ND + c];
    }
    // warm caches for the next K and V chunks
    if (kc + 1 < NCHUNK) {
      __builtin_prefetch(Kp + (size_t)(k0 + BK + pf_row) * ND + pf_half, 0, 3);
      __builtin_prefetch(Vp + (size_t)(k0 + BK + pf_row) * ND + pf_half, 0, 3);
    }
    __syncthreads();

    #pragma unroll
    for (int nt = 0; nt < 4; ++nt) {
      v16h b0, b1;
      #pragma unroll
      for (int j = 0; j < 16; ++j) {
        int kk = j + (j & 8) + 8 * g;
        b0[j] = Ks[(nt * 16 + ln) * LSTR + kk];
        b1[j] = Ks[(nt * 16 + ln) * LSTR + 32 + kk];
      }
      v8f c = vzero;
      c = wmma16(aq0, b0, c);
      c = wmma16(aq1, b1, c);
      #pragma unroll
      for (int r = 0; r < 8; ++r) {
        int m = r + 8 * g;
        int q = q0 + wave * 16 + m;
        int k = k0 + nt * 16 + ln;
        bool ok = ((k + mn) <= q) && (rowok[r] != 0.0f);
        float s = ok ? c[r] * scale : -1.0e9f;
        float p = __expf(s - m_run[r]) * inv_l[r];
        __builtin_nontemporal_store(p, &Pp[(size_t)q * NS + k]);   // 537MB stream: don't pollute L2
        Ps[(wave * 16 + m) * LSTR + nt * 16 + ln] = (_Float16)p;
      }
    }
    __syncthreads();   // Ps C-layout -> A-layout re-read below

    // P A-fragments (16 q rows x 32 keys, two K-steps over the 64-key chunk)
    v16h ap0, ap1;
    #pragma unroll
    for (int j = 0; j < 16; ++j) {
      int kk = j + (j & 8) + 8 * g;
      ap0[j] = Ps[(wave * 16 + ln) * LSTR + kk];
      ap1[j] = Ps[(wave * 16 + ln) * LSTR + 32 + kk];
    }
    #pragma unroll
    for (int nt = 0; nt < 4; ++nt) {
      // B fragment = V (keys x d): half j -> key = j + (j&8) + 8g (+32), n(d) = nt*16+ln
      v16h bv0, bv1;
      #pragma unroll
      for (int j = 0; j < 16; ++j) {
        int kk = j + (j & 8) + 8 * g;
        bv0[j] = Vs[kk * LSTR + nt * 16 + ln];
        bv1[j] = Vs[(32 + kk) * LSTR + nt * 16 + ln];
      }
      acc[nt] = wmma16(ap0, bv0, acc[nt]);
      acc[nt] = wmma16(ap1, bv1, acc[nt]);
    }
  }

  // ---- epilogue: write O tile ----
  #pragma unroll
  for (int nt = 0; nt < 4; ++nt) {
    #pragma unroll
    for (int r = 0; r < 8; ++r) {
      int q = q0 + wave * 16 + r + 8 * g;
      Op[(size_t)q * ND + nt * 16 + ln] = acc[nt][r];
    }
  }
}

extern "C" void kernel_launch(void* const* d_in, const int* in_sizes, int n_in,
                              void* d_out, int out_size, void* d_ws, size_t ws_size,
                              hipStream_t stream) {
  const float* Q  = (const float*)d_in[0];
  const float* K  = (const float*)d_in[1];
  const float* V  = (const float*)d_in[2];
  const float* Mk = (const float*)d_in[3];
  const int*   mn = (const int*)d_in[4];
  float* out   = (float*)d_out;                                  // [B,H,S,D]
  float* probs = out + (size_t)NB * NH * NS * ND;                // [B,H,S,S]
  dim3 grid(NS / BQ, NH, NB);
  attn_fwd<<<grid, dim3(128), 0, stream>>>(Q, K, V, Mk, mn, out, probs);
}
